// SwinTransformerBlock_60576218742984
// MI455X (gfx1250) — matrix operand order repaired and checked by reading, not compiled, not run on hardware
//
#include <hip/hip_runtime.h>

typedef unsigned short u16;
typedef unsigned int   u32;
typedef __attribute__((ext_vector_type(16))) __bf16 v16bf;
typedef __attribute__((ext_vector_type(8)))  float  v8f;

union Frag { v16bf v; u32 u[8]; };

__device__ __forceinline__ u16 f2bf(float f) {
  u32 u = __builtin_bit_cast(u32, f);
  u32 r = u + 0x7FFFu + ((u >> 16) & 1u);
  return (u16)(r >> 16);
}
__device__ __forceinline__ u32 pack2bf(float a, float b) {
  return (u32)f2bf(a) | ((u32)f2bf(b) << 16);
}

__device__ __forceinline__ v8f wmma_bf16(const Frag& a, const Frag& b, v8f c) {
  // D(16x16,f32) = A(16x32,bf16) x B(32x16,bf16) + C
  return __builtin_amdgcn_wmma_f32_16x16x32_bf16(
      /*neg_a=*/false, a.v, /*neg_b=*/false, b.v,
      /*c_mod=*/(short)0, c, /*reuse_a=*/false, /*reuse_b=*/false);
}

// gfx1250 async global->LDS copy (ASYNCcnt-tracked, no VGPR round trip).
// LDS offset = low 32 bits of the generic pointer (generic LDS addr is
// {shared_aperture, offset} per the ISA aperture rules).
__device__ __forceinline__ void async_load_b128(u32 lds_off, const u32* gptr) {
  asm volatile("global_load_async_to_lds_b128 %0, %1, off"
               :: "v"(lds_off), "v"(gptr)
               : "memory");
}
__device__ __forceinline__ void wait_async0() {
  asm volatile("s_wait_asynccnt 0" ::: "memory");
}

// ---------------------------------------------------------------------------
// fp32 -> bf16 weight conversion
// ---------------------------------------------------------------------------
__global__ __launch_bounds__(256) void cvt_kernel(const float* __restrict__ src,
                                                  u16* __restrict__ dst, int n) {
  int i = blockIdx.x * 256 + threadIdx.x;
  if (i < n) dst[i] = f2bf(src[i]);
}

// ---------------------------------------------------------------------------
// LayerNorm (one wave per 192-wide token row, wave32 shuffle reduction).
// windowed=1: gather through roll(-3,-3) + window partition (LN1 path).
// windowed=0: plain token order (LN2 path).
// ---------------------------------------------------------------------------
__global__ __launch_bounds__(256) void ln_kernel(const float* __restrict__ x,
                                                 const float* __restrict__ g,
                                                 const float* __restrict__ bsh,
                                                 u16* __restrict__ out,
                                                 int windowed) {
  int row  = blockIdx.x * 8 + (threadIdx.x >> 5);
  int lane = threadIdx.x & 31;
  int srow = row;
  if (windowed) {
    int win = row / 49, n = row - win * 49;
    int bimg = win >> 6, wrem = win & 63;
    int si = (wrem >> 3) * 7 + n / 7 + 3; if (si >= 56) si -= 56;
    int sj = (wrem & 7) * 7 + n % 7 + 3; if (sj >= 56) sj -= 56;
    srow = bimg * 3136 + si * 56 + sj;
  }
  const float* xr = x + (size_t)srow * 192;
  float vals[6];
  float s = 0.f;
#pragma unroll
  for (int i = 0; i < 6; ++i) { vals[i] = xr[lane + 32 * i]; s += vals[i]; }
#pragma unroll
  for (int m = 16; m > 0; m >>= 1) s += __shfl_xor(s, m, 32);
  float mean = s * (1.0f / 192.0f);
  float vv = 0.f;
#pragma unroll
  for (int i = 0; i < 6; ++i) { float d = vals[i] - mean; vv += d * d; }
#pragma unroll
  for (int m = 16; m > 0; m >>= 1) vv += __shfl_xor(vv, m, 32);
  float inv = rsqrtf(vv * (1.0f / 192.0f) + 1e-5f);
  u16* orow = out + (size_t)row * 192;
#pragma unroll
  for (int i = 0; i < 6; ++i) {
    int c = lane + 32 * i;
    orow[c] = f2bf((vals[i] - mean) * inv * g[c] + bsh[c]);
  }
}

// ---------------------------------------------------------------------------
// Generic bf16 GEMM: C[M,N] = A[M,K] * W[N,K]^T, fp32 accumulate via WMMA.
// 128 threads = 4 waves; block tile 64x64; K-step 32.
// Double-buffered async global->LDS staging: copy of slab i+1 overlaps WMMA
// on slab i; one workgroup barrier per K-step.
// MODE 0: QKV split (out0/1/2 bf16)      MODE 1: proj + unwindow + residual
// MODE 2: exact-GELU -> bf16             MODE 3: residual -> fp32 d_out
// ---------------------------------------------------------------------------
template <int MODE>
__global__ __launch_bounds__(128) void gemm_bf16_kernel(
    const u16* __restrict__ A, const u16* __restrict__ Bw,
    const float* __restrict__ bias, int M, int N, int K,
    u16* __restrict__ out0, u16* __restrict__ out1, u16* __restrict__ out2,
    float* __restrict__ outf, const float* __restrict__ resid) {
  __shared__ __attribute__((aligned(16))) u32 As[2][64][20];  // [buf][m][k/2]
  __shared__ __attribute__((aligned(16))) u32 Bs[2][64][20];  // [buf][n][k/2]
  const int tid  = threadIdx.x;
  const int lane = tid & 31;
  const int wave = tid >> 5;
  const int tileM = blockIdx.y * 64;
  const int tileN = blockIdx.x * 64;
  const u32* Au = (const u32*)A;
  const u32* Bu = (const u32*)Bw;
  const int Ku = K >> 1;

  // each thread owns two 16B chunks per matrix per slab
  const int ca = tid, cb = tid + 128;            // chunk ids 0..255
  const int ra = ca >> 2, ua = (ca & 3) << 2;    // row, uint-offset
  const int rb = cb >> 2, ub = (cb & 3) << 2;

  v8f acc[4] = {};

  const int nIter = K >> 5;
  // prime slab 0 into buffer 0
  {
    const int ku0 = 0;
    async_load_b128((u32)(size_t)&As[0][ra][ua], Au + (size_t)(tileM + ra) * Ku + ku0 + ua);
    async_load_b128((u32)(size_t)&As[0][rb][ub], Au + (size_t)(tileM + rb) * Ku + ku0 + ub);
    async_load_b128((u32)(size_t)&Bs[0][ra][ua], Bu + (size_t)(tileN + ra) * Ku + ku0 + ua);
    async_load_b128((u32)(size_t)&Bs[0][rb][ub], Bu + (size_t)(tileN + rb) * Ku + ku0 + ub);
  }

  for (int it = 0; it < nIter; ++it) {
    const int buf = it & 1;
    wait_async0();          // this wave's pending slab is in LDS
    __syncthreads();        // all waves' slabs visible; prev buf fully consumed
    if (it + 1 < nIter) {   // overlap: fetch next slab into the other buffer
      const int ku0 = (it + 1) << 4;
      const int nb = buf ^ 1;
      async_load_b128((u32)(size_t)&As[nb][ra][ua], Au + (size_t)(tileM + ra) * Ku + ku0 + ua);
      async_load_b128((u32)(size_t)&As[nb][rb][ub], Au + (size_t)(tileM + rb) * Ku + ku0 + ub);
      async_load_b128((u32)(size_t)&Bs[nb][ra][ua], Bu + (size_t)(tileN + ra) * Ku + ku0 + ua);
      async_load_b128((u32)(size_t)&Bs[nb][rb][ub], Bu + (size_t)(tileN + rb) * Ku + ku0 + ub);
    }

    Frag a;  // A frag: u[v] = As[m][(v>=4?8:0)+(lane>=16?4:0)+(v&3)]
    {
      const u32* ap = &As[buf][wave * 16 + (lane & 15)][(lane >= 16) ? 4 : 0];
#pragma unroll
      for (int v = 0; v < 4; ++v) { a.u[v] = ap[v]; a.u[v + 4] = ap[v + 8]; }
    }
#pragma unroll
    for (int nt = 0; nt < 4; ++nt) {  // B frag: u[v] = Bs[n][(v>=4?8:0)+(lane>=16?4:0)+(v&3)]
      Frag b;
      const u32* bp = &Bs[buf][nt * 16 + (lane & 15)][(lane >= 16) ? 4 : 0];
#pragma unroll
      for (int v = 0; v < 4; ++v) { b.u[v] = bp[v]; b.u[v + 4] = bp[v + 8]; }
      acc[nt] = wmma_bf16(a, b, acc[nt]);
    }
  }

  const int mBase = wave * 16 + ((lane >= 16) ? 8 : 0);
  const int nLane = lane & 15;
#pragma unroll
  for (int nt = 0; nt < 4; ++nt) {
#pragma unroll
    for (int v = 0; v < 8; ++v) {
      int row = tileM + mBase + v;
      int col = tileN + nt * 16 + nLane;
      float val = acc[nt][v] + bias[col];
      if (MODE == 0) {
        int which = col / 192, c = col - which * 192;
        u16* dst = (which == 0) ? out0 : (which == 1) ? out1 : out2;
        dst[(size_t)row * 192 + c] = f2bf(val);
      } else if (MODE == 1) {
        int win = row / 49, n = row - win * 49;
        int bimg = win >> 6, wrem = win & 63;
        int si = (wrem >> 3) * 7 + n / 7 + 3; if (si >= 56) si -= 56;
        int sj = (wrem & 7) * 7 + n % 7 + 3; if (sj >= 56) sj -= 56;
        size_t tok = (size_t)bimg * 3136 + (size_t)si * 56 + sj;
        outf[tok * 192 + col] = resid[tok * 192 + col] + val;
      } else if (MODE == 2) {
        float gl = 0.5f * val * (1.0f + erff(val * 0.70710678118654752f));
        out0[(size_t)row * 768 + col] = f2bf(gl);
      } else {
        size_t o = (size_t)row * 192 + col;
        outf[o] = resid[o] + val;
      }
    }
  }
}

// ---------------------------------------------------------------------------
// Windowed attention: one block per (window, head).  49 tokens padded to 64.
// S = Q*K^T (WMMA), + scale/rpb/mask, softmax, O = P*V (WMMA).
// ---------------------------------------------------------------------------
__global__ __launch_bounds__(128) void swin_attn_kernel(
    const u16* __restrict__ Q, const u16* __restrict__ Kx,
    const u16* __restrict__ V, const float* __restrict__ rpb,
    const float* __restrict__ mask, u16* __restrict__ O) {
  const int win  = blockIdx.x;
  const int head = blockIdx.y;
  const int tid  = threadIdx.x;
  const int lane = tid & 31;
  const int wave = tid >> 5;

  __shared__ __attribute__((aligned(16))) u32 Qs[64][20];    // [tok][d/2]
  __shared__ __attribute__((aligned(16))) u32 Ks[64][20];    // [key][d/2]
  __shared__ __attribute__((aligned(16))) u32 Vtu[32][36];   // V^T: [dim][key/2]
  __shared__ __attribute__((aligned(16))) float Sf[64][65];  // logits / probs
  __shared__ __attribute__((aligned(16))) u32 Pu[64][36];    // bf16 probs [tok][key/2]

  const u32* Qu = (const u32*)Q;
  const u32* Ku = (const u32*)Kx;
  const size_t rb = (size_t)win * 49;

  for (int idx = tid; idx < 64 * 16; idx += 128) {
    int r = idx >> 4, c = idx & 15;
    u32 qv = 0, kv = 0;
    if (r < 49) {
      size_t gofs = (rb + r) * 96 + head * 16 + c;
      qv = Qu[gofs];
      kv = Ku[gofs];
    }
    Qs[r][c] = qv;
    Ks[r][c] = kv;
  }
  u16* Vts = (u16*)Vtu;  // row stride 72 ushorts
  for (int idx = tid; idx < 64 * 32; idx += 128) {
    int key = idx >> 5, d = idx & 31;
    u16 vv = 0;
    if (key < 49) vv = V[(rb + key) * 192 + head * 32 + d];
    Vts[d * 72 + key] = vv;
  }
  __syncthreads();

  // ---- S = Q * K^T  (each wave owns one 16-row m-tile) ----
  {
    Frag a;
    const u32* ap = &Qs[wave * 16 + (lane & 15)][(lane >= 16) ? 4 : 0];
#pragma unroll
    for (int v = 0; v < 4; ++v) { a.u[v] = ap[v]; a.u[v + 4] = ap[v + 8]; }
    const int mB = wave * 16 + ((lane >= 16) ? 8 : 0);
#pragma unroll
    for (int nt = 0; nt < 4; ++nt) {
      Frag b;
      const u32* bp = &Ks[nt * 16 + (lane & 15)][(lane >= 16) ? 4 : 0];
#pragma unroll
      for (int v = 0; v < 4; ++v) { b.u[v] = bp[v]; b.u[v + 4] = bp[v + 8]; }
      v8f c = {};
      c = wmma_bf16(a, b, c);
#pragma unroll
      for (int v = 0; v < 8; ++v) Sf[mB + v][nt * 16 + (lane & 15)] = c[v];
    }
  }
  __syncthreads();

  // ---- scale + rpb + shift-mask + softmax, write bf16 probs ----
  if (tid < 64) {
    int t = tid;
    if (t < 49) {
      const float scale = 0.17677669529663687f;  // 1/sqrt(32)
      int qi = t / 7, qj = t - qi * 7;
      const float* mrow = mask + ((size_t)(win & 63) * 49 + t) * 49;
      float mx = -1e30f;
      for (int j = 0; j < 49; ++j) {
        int ki = j / 7, kj = j - ki * 7;
        int ridx = (qi - ki + 6) * 13 + (qj - kj + 6);
        float s = Sf[t][j] * scale + rpb[ridx * 6 + head] + mrow[j];
        Sf[t][j] = s;
        mx = fmaxf(mx, s);
      }
      float sum = 0.f;
      for (int j = 0; j < 49; ++j) {
        float e = __expf(Sf[t][j] - mx);
        sum += e;
        Sf[t][j] = e;
      }
      float inv = 1.0f / sum;
      for (int jp = 0; jp < 32; ++jp) {
        int j0 = jp * 2, j1 = j0 + 1;
        float f0 = (j0 < 49) ? Sf[t][j0] * inv : 0.f;
        float f1 = (j1 < 49) ? Sf[t][j1] * inv : 0.f;
        Pu[t][jp] = pack2bf(f0, f1);
      }
    } else {
      for (int jp = 0; jp < 32; ++jp) Pu[t][jp] = 0u;
    }
  }
  __syncthreads();

  // ---- O = P * V  (K-dim 64 -> two WMMA steps; N=32 -> two n-tiles) ----
  v8f acc2[2] = {};
#pragma unroll
  for (int k0 = 0; k0 < 64; k0 += 32) {
    Frag a;
    const u32* ap = &Pu[wave * 16 + (lane & 15)][(k0 >> 1) + ((lane >= 16) ? 4 : 0)];
#pragma unroll
    for (int v = 0; v < 4; ++v) { a.u[v] = ap[v]; a.u[v + 4] = ap[v + 8]; }
#pragma unroll
    for (int nt = 0; nt < 2; ++nt) {
      Frag b;
      const u32* bp = &Vtu[nt * 16 + (lane & 15)][(k0 >> 1) + ((lane >= 16) ? 4 : 0)];
#pragma unroll
      for (int v = 0; v < 4; ++v) { b.u[v] = bp[v]; b.u[v + 4] = bp[v + 8]; }
      acc2[nt] = wmma_bf16(a, b, acc2[nt]);
    }
  }
  const int mB = wave * 16 + ((lane >= 16) ? 8 : 0);
#pragma unroll
  for (int nt = 0; nt < 2; ++nt) {
#pragma unroll
    for (int v = 0; v < 8; ++v) {
      int m = mB + v;
      if (m < 49)
        O[(rb + m) * 192 + head * 32 + nt * 16 + (lane & 15)] = f2bf(acc2[nt][v]);
    }
  }
}

// ---------------------------------------------------------------------------
// Host-side orchestration
// ---------------------------------------------------------------------------
extern "C" void kernel_launch(void* const* d_in, const int* in_sizes, int n_in,
                              void* d_out, int out_size, void* d_ws, size_t ws_size,
                              hipStream_t stream) {
  const float* x      = (const float*)d_in[0];
  const float* maskm  = (const float*)d_in[1];
  const float* n1g    = (const float*)d_in[2];
  const float* n1b    = (const float*)d_in[3];
  const float* qkv_w  = (const float*)d_in[4];
  const float* qkv_b  = (const float*)d_in[5];
  const float* rpb    = (const float*)d_in[6];
  const float* proj_w = (const float*)d_in[7];
  const float* proj_b = (const float*)d_in[8];
  const float* n2g    = (const float*)d_in[9];
  const float* n2b    = (const float*)d_in[10];
  const float* fc1_w  = (const float*)d_in[11];
  const float* fc1_b  = (const float*)d_in[12];
  const float* fc2_w  = (const float*)d_in[13];
  const float* fc2_b  = (const float*)d_in[14];

  // 100352 rows (= 32 imgs * 64 windows * 49 tokens), C=192, HID=768.
  const size_t ROWS = 100352;
  const size_t SZ_BF192 = ROWS * 192 * 2;       // 38,535,168 B
  char* ws = (char*)d_ws;
  u16*   W0   = (u16*)(ws);                     // LN1 windows, later attn out
  u16*   Qb   = (u16*)(ws + 1 * SZ_BF192);      // also reused as LN2 output
  u16*   Kb   = (u16*)(ws + 2 * SZ_BF192);
  u16*   Vb   = (u16*)(ws + 3 * SZ_BF192);
  float* X2   = (float*)(ws + 4 * SZ_BF192);    // x + attn branch (fp32)
  u16*   F1   = (u16*)(ws + 4 * SZ_BF192 + ROWS * 192 * 4);
  char*  wsm  = ws + 4 * SZ_BF192 + ROWS * 192 * 4 + ROWS * 768 * 2;
  u16*   WQKV  = (u16*)(wsm);
  u16*   WPROJ = (u16*)(wsm + 576 * 192 * 2);
  u16*   WFC1  = (u16*)(wsm + 576 * 192 * 2 + 192 * 192 * 2);
  u16*   WFC2  = (u16*)(wsm + 576 * 192 * 2 + 192 * 192 * 2 + 768 * 192 * 2);
  // total workspace use ~386 MB

  // 1) weights fp32 -> bf16
  cvt_kernel<<<(110592 + 255) / 256, 256, 0, stream>>>(qkv_w, WQKV, 110592);
  cvt_kernel<<<(36864 + 255) / 256, 256, 0, stream>>>(proj_w, WPROJ, 36864);
  cvt_kernel<<<(147456 + 255) / 256, 256, 0, stream>>>(fc1_w, WFC1, 147456);
  cvt_kernel<<<(147456 + 255) / 256, 256, 0, stream>>>(fc2_w, WFC2, 147456);

  // 2) LN1 + roll(-3,-3) + window partition -> bf16
  ln_kernel<<<12544, 256, 0, stream>>>(x, n1g, n1b, W0, 1);

  // 3) QKV GEMM: [100352,192] x [576,192]^T
  gemm_bf16_kernel<0><<<dim3(9, 1568), 128, 0, stream>>>(
      W0, WQKV, qkv_b, 100352, 576, 192, Qb, Kb, Vb, nullptr, nullptr);

  // 4) windowed attention (2048 windows x 6 heads); output reuses W0
  swin_attn_kernel<<<dim3(2048, 6), 128, 0, stream>>>(Qb, Kb, Vb, rpb, maskm, W0);

  // 5) proj GEMM + window-reverse + roll(+3,+3) + residual -> X2 (fp32)
  gemm_bf16_kernel<1><<<dim3(3, 1568), 128, 0, stream>>>(
      W0, WPROJ, proj_b, 100352, 192, 192, nullptr, nullptr, nullptr, X2, x);

  // 6) LN2 -> bf16 (reuse Qb)
  ln_kernel<<<12544, 256, 0, stream>>>(X2, n2g, n2b, Qb, 0);

  // 7) fc1 GEMM + exact GELU -> bf16
  gemm_bf16_kernel<2><<<dim3(12, 1568), 128, 0, stream>>>(
      Qb, WFC1, fc1_b, 100352, 768, 192, F1, nullptr, nullptr, nullptr, nullptr);

  // 8) fc2 GEMM + residual -> d_out (fp32)
  gemm_bf16_kernel<3><<<dim3(3, 1568), 128, 0, stream>>>(
      F1, WFC2, fc2_b, 100352, 192, 768, nullptr, nullptr, nullptr,
      (float*)d_out, X2);
}
